// LSTM_21105469292942
// MI455X (gfx1250) — matrix operand order repaired and checked by reading, not compiled, not run on hardware
//
#include <hip/hip_runtime.h>
#include <hip/hip_bf16.h>

// ---------------------------------------------------------------------------
// 3-layer LSTM recurrence on MI455X (gfx1250, wave32, WMMA).
//
// Decomposition: grid = 32 WGs = 8 batch-groups (16 rows each, WMMA M=16)
//                x 4 gate-aligned column-groups (128 hidden units each).
// Batch groups are fully independent (no sync). Column groups of one batch
// group rendezvous via a 4-member L2 spin barrier 3x per timestep; hidden
// state h is double-buffered by timestep parity so the rendezvous is
// race-free. Cell state c lives in LDS for the entire 1023-step loop.
// GEMMs use v_wmma_f32_16x16x32_f16 with f32 accumulation; combined
// [w_ih | w_hh] weights are pre-converted to row-major f16 (row n of W is
// exactly the contiguous B-fragment for output column n -- no transpose).
// A-staging uses GLOBAL_LOAD_ASYNC_TO_LDS_B128 (ASYNCcnt) when available.
// ---------------------------------------------------------------------------

#define B_   128
#define T_   1024
#define IN_  256
#define H_   512
#define G4_  2048
#define WGT  512     // threads per WG = 16 waves (wave32)

typedef __attribute__((ext_vector_type(16))) _Float16 v16h;
typedef __attribute__((ext_vector_type(8)))  _Float16 h8v;
typedef __attribute__((ext_vector_type(4)))  _Float16 h4v;
typedef __attribute__((ext_vector_type(8)))  float    v8f;
typedef __attribute__((ext_vector_type(4)))  int      v4i;

// Workspace layout (bytes). Requires ws_size >= ~12.7 MB.
#define WS_WC1   0                               // f16 [2048][768]  = [w_ih1|w_hh1]
#define WS_WC2   (WS_WC1 + 2048*768*2)           // f16 [2048][1024] = [w_ih2|w_hh2]
#define WS_WC3   (WS_WC2 + 2048*1024*2)          // f16 [2048][1024]
#define WS_FCW   (WS_WC3 + 2048*1024*2)          // f16 [256][512]
#define WS_B1    (WS_FCW + 256*512*2)            // f32 [2048] b_ih1+b_hh1
#define WS_B2    (WS_B1 + 2048*4)
#define WS_B3    (WS_B2 + 2048*4)
#define WS_FCB   (WS_B3 + 2048*4)                // f32 [256]
#define WS_H     (WS_FCB + 256*4)                // f16 [3][2][128][512] (dbl-buffered h)
#define WS_BARS  (WS_H + 3*2*128*512*2)          // u32 [8][64] barrier counters
#define WS_TOTAL (WS_BARS + 8*64*4)

// --- CDNA5 async global->LDS copy (ASYNCcnt), with safe fallback ------------
#if defined(__has_builtin)
#if __has_builtin(__builtin_amdgcn_global_load_async_to_lds_b128) && \
    __has_builtin(__builtin_amdgcn_s_wait_asynccnt)
#define USE_ASYNC_LDS 1
#endif
#endif
#ifndef USE_ASYNC_LDS
#define USE_ASYNC_LDS 0
#endif

// Copy 16 aligned bytes from global to LDS. Async path: no VGPR round-trip,
// tracked with ASYNCcnt (global_load_async_to_lds_b128, ISA 10.x / 15.18.3).
// Builtin prototype is (int4 AS1* src, int4 AS3* dst, imm offset, imm cpol).
__device__ __forceinline__ void copy16_g2lds(const _Float16* g, _Float16* l) {
#if USE_ASYNC_LDS
  __builtin_amdgcn_global_load_async_to_lds_b128(
      (__attribute__((address_space(1))) v4i*)g,
      (__attribute__((address_space(3))) v4i*)l, 0, 0);
#else
  *(uint4*)l = *(const uint4*)g;
#endif
}
__device__ __forceinline__ void async_join() {
#if USE_ASYNC_LDS
  __builtin_amdgcn_s_wait_asynccnt(0);   // s_wait_asynccnt 0
#endif
}

// --- WMMA fragment builders (layouts per CDNA5 ISA 7.12.2) ------------------
__device__ __forceinline__ v16h make_frag(const _Float16* p0, const _Float16* p1) {
  h8v lo = *(const h8v*)p0;
  h8v hi = *(const h8v*)p1;
  v16h r;
#pragma unroll
  for (int i = 0; i < 8; ++i) { r[i] = lo[i]; r[i + 8] = hi[i]; }
  return r;
}
// A 16x32 f16: lane (m = lane&15) holds row m; halves[0:8] = K[kk+8*lh .. +8),
// halves[8:16] = K[kk+16+8*lh .. +8), lh = lane>>4.
__device__ __forceinline__ v16h a_frag(const _Float16* a_row, int kk, int lh) {
  return make_frag(a_row + kk + 8 * lh, a_row + kk + 16 + 8 * lh);
}
// B 32x16 f16: lane holds column n = lane&15, halves[0:8] = K[kk+8*lh .. +8),
// halves[8:16] = K[kk+16+8*lh .. +8), from row n of the row-major weight matrix W ( == column n of B = W^T ).
__device__ __forceinline__ v16h b_frag(const _Float16* w_row, int kk, int lh) {
  const _Float16* p = w_row + kk + 8 * lh;
  return make_frag(p, p + 16);
}

__device__ __forceinline__ float sigmoid_f(float x) { return 1.0f / (1.0f + __expf(-x)); }
__device__ __forceinline__ float tanh_f(float x)    { return 1.0f - 2.0f / (__expf(2.0f * x) + 1.0f); }

// 4-member inter-WG rendezvous through L2 (monotonic, phase-free counter).
__device__ __forceinline__ void group_barrier(unsigned* bar) {
  __threadfence();                 // publish h stores at device scope
  __syncthreads();
  if (threadIdx.x == 0) {
    unsigned old = __hip_atomic_fetch_add(bar, 1u, __ATOMIC_ACQ_REL, __HIP_MEMORY_SCOPE_AGENT);
    unsigned target = (old & ~3u) + 4u;
    while (__hip_atomic_load(bar, __ATOMIC_ACQUIRE, __HIP_MEMORY_SCOPE_AGENT) < target)
      __builtin_amdgcn_s_sleep(2);
  }
  __syncthreads();
  __threadfence();                 // acquire other groups' h stores
}

// --- one LSTM cell for this WG's 16 batch rows x 128 hidden units -----------
// Albs/gl alias the same 32KB LDS union: A (f16 [16][KTOT]) then gates (f32 [16][512]).
template <int KTOT>
__device__ __forceinline__ void lstm_cell_step(
    _Float16* Albs, float* gl, float* cslice,
    const _Float16* __restrict__ Wc, const float* __restrict__ bias,
    _Float16* __restrict__ hout, int r0, int cg, unsigned* bar)
{
  const int tid  = threadIdx.x;
  const int wave = tid >> 5, lane = tid & 31;
  const int nlo  = lane & 15, lh = lane >> 4;
  const int tt0  = wave * 2;                       // 2 of 32 16x16 tiles per wave
  const int n0a  = (tt0 >> 3) * H_ + cg * 128 + (tt0 & 7) * 16;
  const int n0b  = n0a + 16;                       // tt0+1 is always in the same gate

  v8f acc0, acc1;
  { // C init with bias (bias depends only on the N column = nlo within tile)
    float ba = bias[n0a + nlo], bb = bias[n0b + nlo];
#pragma unroll
    for (int i = 0; i < 8; ++i) { acc0[i] = ba; acc1[i] = bb; }
  }
  const _Float16* arow = Albs + (size_t)nlo * KTOT;
  const _Float16* w0   = Wc + (size_t)(n0a + nlo) * KTOT;
  const _Float16* w1   = Wc + (size_t)(n0b + nlo) * KTOT;
#pragma unroll 4
  for (int kk = 0; kk < KTOT; kk += 32) {
    v16h a  = a_frag(arow, kk, lh);                // shared by both tiles
    v16h b0 = b_frag(w0, kk, lh);
    v16h b1 = b_frag(w1, kk, lh);
    acc0 = __builtin_amdgcn_wmma_f32_16x16x32_f16(false, a, false, b0, (short)0, acc0, false, false);
    acc1 = __builtin_amdgcn_wmma_f32_16x16x32_f16(false, a, false, b1, (short)0, acc1, false, false);
  }
  __syncthreads();                                  // done reading A -> reuse union as gates
#pragma unroll
  for (int v = 0; v < 8; ++v) {                     // C/D layout: row = v + 8*lh, col = nlo
    int row = v + 8 * lh;
    gl[row * H_ + tt0 * 16 + nlo]       = acc0[v];
    gl[row * H_ + (tt0 + 1) * 16 + nlo] = acc1[v];
  }
  __syncthreads();
  // elementwise cell update; local col q*128+hj holds gate q for hidden unit hj.
  // Process hj in pairs so h goes out as one packed 2xf16 u32 store.
  for (int e = tid; e < 16 * 64; e += WGT) {
    int row = e >> 6, hp = (e & 63) * 2;
    union { _Float16 h[2]; unsigned u; } pk;
#pragma unroll
    for (int s = 0; s < 2; ++s) {
      int hj = hp + s;
      float gi = gl[row * H_ + 0 * 128 + hj];
      float gf = gl[row * H_ + 1 * 128 + hj];
      float gg = gl[row * H_ + 2 * 128 + hj];
      float go = gl[row * H_ + 3 * 128 + hj];
      float c  = sigmoid_f(gf) * cslice[row * 128 + hj] + sigmoid_f(gi) * tanh_f(gg);
      cslice[row * 128 + hj] = c;
      pk.h[s] = (_Float16)(sigmoid_f(go) * tanh_f(c));
    }
    *(unsigned*)&hout[(size_t)(r0 + row) * H_ + cg * 128 + hp] = pk.u;
  }
  group_barrier(bar);                               // h slice published to sibling WGs
}

// Stage a [16][512]-f16 block of h (rows r0..r0+15) into LDS rows of stride
// STRIDE halves at column colOff. 16B chunks; async when available.
template <int STRIDE>
__device__ __forceinline__ void stage_h(const _Float16* __restrict__ src,
                                        _Float16* Albs, int r0, int colOff) {
  for (int c = threadIdx.x; c < 16 * 64; c += WGT) {
    int row = c >> 6, q = (c & 63) * 8;
    copy16_g2lds(src + (size_t)(r0 + row) * H_ + q, Albs + row * STRIDE + colOff + q);
  }
}

__global__ __launch_bounds__(WGT, 1) void lstm_persistent(
    const float* __restrict__ embed,
    const _Float16* __restrict__ Wc1, const _Float16* __restrict__ Wc2,
    const _Float16* __restrict__ Wc3,
    const float* __restrict__ b1, const float* __restrict__ b2, const float* __restrict__ b3,
    const _Float16* __restrict__ fcW, const float* __restrict__ fcb,
    _Float16* __restrict__ hbuf, unsigned* __restrict__ bars,
    float* __restrict__ out)
{
  __shared__ float s_ag[16 * H_];        // 32KB union: A-staging (f16) / gates (f32)
  __shared__ float s_c[3 * 16 * 128];    // 24KB persistent cell state (3 layers)
  _Float16* Albs = (_Float16*)s_ag;
  float*    gl   = s_ag;

  const int tid = threadIdx.x;
  const int bg  = blockIdx.x >> 2;       // batch group 0..7
  const int cg  = blockIdx.x & 3;        // column group 0..3
  const int r0  = bg * 16;
  unsigned* bar = bars + bg * 64;

  for (int e = tid; e < 3 * 16 * 128; e += WGT) s_c[e] = 0.0f;
  __syncthreads();

  const size_t HB = (size_t)B_ * H_;
  _Float16* h1 = hbuf;
  _Float16* h2 = hbuf + 2 * HB;
  _Float16* h3 = hbuf + 4 * HB;

  for (int t = 0; t < T_ - 1; ++t) {
    const int pc = t & 1, pp = pc ^ 1;
    // ---- layer 1: A = [ embed[:, t, :] | h1(t-1) ], K = 768
    // embed needs f32->f16 conversion: float4 load -> packed 4xf16 ds_store_b64
    for (int c = tid; c < 16 * (IN_ / 4); c += WGT) {
      int row = c >> 6, q = (c & 63) * 4;
      float4 v = *(const float4*)(embed + ((size_t)(r0 + row) * T_ + t) * IN_ + q);
      h4v hv = { (_Float16)v.x, (_Float16)v.y, (_Float16)v.z, (_Float16)v.w };
      *(h4v*)(Albs + row * 768 + q) = hv;
    }
    stage_h<768>(h1 + (size_t)pp * HB, Albs, r0, IN_);
    async_join();
    __syncthreads();
    lstm_cell_step<768>(Albs, gl, s_c, Wc1, b1, h1 + (size_t)pc * HB, r0, cg, bar);

    // ---- layer 2: A = [ h1(t) | h2(t-1) ], K = 1024
    stage_h<1024>(h1 + (size_t)pc * HB, Albs, r0, 0);
    stage_h<1024>(h2 + (size_t)pp * HB, Albs, r0, H_);
    async_join();
    __syncthreads();
    lstm_cell_step<1024>(Albs, gl, s_c + 16 * 128, Wc2, b2, h2 + (size_t)pc * HB, r0, cg, bar);

    // ---- layer 3: A = [ h2(t) | h3(t-1) ], K = 1024
    stage_h<1024>(h2 + (size_t)pc * HB, Albs, r0, 0);
    stage_h<1024>(h3 + (size_t)pp * HB, Albs, r0, H_);
    async_join();
    __syncthreads();
    lstm_cell_step<1024>(Albs, gl, s_c + 2 * 16 * 128, Wc3, b3, h3 + (size_t)pc * HB, r0, cg, bar);
  }

  // ---- fc on final h3: out[16 rows][cg*64 .. cg*64+64), K = 512
  {
    const _Float16* h3f = h3 + (size_t)((T_ - 2) & 1) * HB;
    stage_h<512>(h3f, Albs, r0, 0);
    async_join();
    __syncthreads();
    const int wave = tid >> 5, lane = tid & 31;
    const int nlo = lane & 15, lh = lane >> 4;
    if (wave < 4) {                                  // wave-uniform: EXEC all-ones inside
      const int n0 = cg * 64 + wave * 16;
      v8f acc;
      float bv = fcb[n0 + nlo];
#pragma unroll
      for (int i = 0; i < 8; ++i) acc[i] = bv;
      const _Float16* arow = Albs + (size_t)nlo * H_;
      const _Float16* wrow = fcW + (size_t)(n0 + nlo) * H_;
#pragma unroll 4
      for (int kk = 0; kk < H_; kk += 32) {
        v16h a = a_frag(arow, kk, lh);
        v16h b = b_frag(wrow, kk, lh);
        acc = __builtin_amdgcn_wmma_f32_16x16x32_f16(false, a, false, b, (short)0, acc, false, false);
      }
#pragma unroll
      for (int v = 0; v < 8; ++v)
        out[(size_t)(r0 + v + 8 * lh) * IN_ + n0 + nlo] = acc[v];
    }
  }
}

// --- embed passthrough: first tuple element of the reference output ---------
__global__ void copy_embed(const float4* __restrict__ src, float4* __restrict__ dst, size_t n4) {
  size_t stride = (size_t)gridDim.x * blockDim.x;
  for (size_t i = (size_t)blockIdx.x * blockDim.x + threadIdx.x; i < n4; i += stride)
    dst[i] = src[i];
}

// --- weight/bias prep: f16 [w_ih|w_hh] concat, bias fold, state zeroing -----
__global__ void prep_kernel(
    const float* __restrict__ w_ih1, const float* __restrict__ w_hh1,
    const float* __restrict__ b_ih1, const float* __restrict__ b_hh1,
    const float* __restrict__ w_ih2, const float* __restrict__ w_hh2,
    const float* __restrict__ b_ih2, const float* __restrict__ b_hh2,
    const float* __restrict__ w_ih3, const float* __restrict__ w_hh3,
    const float* __restrict__ b_ih3, const float* __restrict__ b_hh3,
    const float* __restrict__ fc_w, const float* __restrict__ fc_b,
    _Float16* __restrict__ Wc1, _Float16* __restrict__ Wc2, _Float16* __restrict__ Wc3,
    _Float16* __restrict__ fcW,
    float* __restrict__ b1, float* __restrict__ b2, float* __restrict__ b3,
    float* __restrict__ fcb,
    _Float16* __restrict__ hbuf, unsigned* __restrict__ bars)
{
  size_t tid = (size_t)blockIdx.x * blockDim.x + threadIdx.x;
  size_t stride = (size_t)gridDim.x * blockDim.x;
  for (size_t e = tid; e < (size_t)2048 * 768; e += stride) {
    int n = (int)(e / 768), k = (int)(e % 768);
    float v = (k < IN_) ? w_ih1[(size_t)n * IN_ + k] : w_hh1[(size_t)n * H_ + (k - IN_)];
    Wc1[e] = (_Float16)v;
  }
  for (size_t e = tid; e < (size_t)2048 * 1024; e += stride) {
    int n = (int)(e >> 10), k = (int)(e & 1023);
    float v = (k < H_) ? w_ih2[(size_t)n * H_ + k] : w_hh2[(size_t)n * H_ + (k - H_)];
    Wc2[e] = (_Float16)v;
  }
  for (size_t e = tid; e < (size_t)2048 * 1024; e += stride) {
    int n = (int)(e >> 10), k = (int)(e & 1023);
    float v = (k < H_) ? w_ih3[(size_t)n * H_ + k] : w_hh3[(size_t)n * H_ + (k - H_)];
    Wc3[e] = (_Float16)v;
  }
  for (size_t e = tid; e < (size_t)IN_ * H_; e += stride) fcW[e] = (_Float16)fc_w[e];
  for (size_t e = tid; e < G4_; e += stride) {
    b1[e] = b_ih1[e] + b_hh1[e];
    b2[e] = b_ih2[e] + b_hh2[e];
    b3[e] = b_ih3[e] + b_hh3[e];
  }
  for (size_t e = tid; e < IN_; e += stride) fcb[e] = fc_b[e];
  for (size_t e = tid; e < (size_t)3 * 2 * B_ * H_; e += stride) hbuf[e] = (_Float16)0.0f;
  for (size_t e = tid; e < 8 * 64; e += stride) bars[e] = 0u;
}

extern "C" void kernel_launch(void* const* d_in, const int* in_sizes, int n_in,
                              void* d_out, int out_size, void* d_ws, size_t ws_size,
                              hipStream_t stream) {
  (void)in_sizes; (void)n_in; (void)out_size; (void)ws_size; // needs ws_size >= WS_TOTAL (~12.7MB)
  const float* embed = (const float*)d_in[0];
  const float* w_ih1 = (const float*)d_in[1];  const float* w_hh1 = (const float*)d_in[2];
  const float* b_ih1 = (const float*)d_in[3];  const float* b_hh1 = (const float*)d_in[4];
  const float* w_ih2 = (const float*)d_in[5];  const float* w_hh2 = (const float*)d_in[6];
  const float* b_ih2 = (const float*)d_in[7];  const float* b_hh2 = (const float*)d_in[8];
  const float* w_ih3 = (const float*)d_in[9];  const float* w_hh3 = (const float*)d_in[10];
  const float* b_ih3 = (const float*)d_in[11]; const float* b_hh3 = (const float*)d_in[12];
  const float* fc_w  = (const float*)d_in[13]; const float* fc_b  = (const float*)d_in[14];

  char* ws = (char*)d_ws;
  _Float16* Wc1 = (_Float16*)(ws + WS_WC1);
  _Float16* Wc2 = (_Float16*)(ws + WS_WC2);
  _Float16* Wc3 = (_Float16*)(ws + WS_WC3);
  _Float16* fcW = (_Float16*)(ws + WS_FCW);
  float* b1  = (float*)(ws + WS_B1);
  float* b2  = (float*)(ws + WS_B2);
  float* b3  = (float*)(ws + WS_B3);
  float* fcb = (float*)(ws + WS_FCB);
  _Float16* hbuf = (_Float16*)(ws + WS_H);
  unsigned* bars = (unsigned*)(ws + WS_BARS);

  float* out_all = (float*)d_out;
  float* out_fc  = out_all + (size_t)B_ * T_ * IN_;

  // 1) embed -> d_out[0 : B*T*IN] (pure bandwidth, ~11us at 23.3 TB/s)
  size_t n4 = (size_t)B_ * T_ * IN_ / 4;
  copy_embed<<<4096, 256, 0, stream>>>((const float4*)embed, (float4*)out_all, n4);

  // 2) weight prep / state init (recomputed every call: no cross-call state)
  prep_kernel<<<2048, 256, 0, stream>>>(
      w_ih1, w_hh1, b_ih1, b_hh1, w_ih2, w_hh2, b_ih2, b_hh2,
      w_ih3, w_hh3, b_ih3, b_hh3, fc_w, fc_b,
      Wc1, Wc2, Wc3, fcW, b1, b2, b3, fcb, hbuf, bars);

  // 3) persistent recurrence: 32 WGs (8 batch x 4 column groups), 16 waves each
  lstm_persistent<<<32, WGT, 0, stream>>>(
      embed, Wc1, Wc2, Wc3, b1, b2, b3, fcW, fcb, hbuf, bars, out_fc);
}
